// AttentiveConv_7576322310835
// MI455X (gfx1250) — matrix operand order repaired and checked by reading, not compiled, not run on hardware
//
#include <hip/hip_runtime.h>
#include <hip/hip_bf16.h>

// ---------------------------------------------------------------------------
// AttentiveConv for MI455X (gfx1250, wave32, WMMA).
//   out = ((softmax_axis1((text@We) @ ctx^T)) @ ctx) @ W2^T
// B=8, N=2048, M=2048, D=1024.  All GEMMs run as bf16 WMMA (f32 accum).
// All GEMMs use B stored as [cols x K] row-major (so B staging is b128-clean);
// We^T and ctx^T are materialized once by an LDS-tiled transpose kernel.
// Workspace requirement: ~239 MB (layout in kernel_launch).
// ---------------------------------------------------------------------------

typedef __attribute__((ext_vector_type(16))) __bf16 v16bf;
typedef __attribute__((ext_vector_type(8)))  float  v8f;
typedef __attribute__((ext_vector_type(4)))  __bf16 v4bf;

// ------------------------------ fp32 -> bf16 -------------------------------
__global__ void f32_to_bf16_kernel(const float* __restrict__ in,
                                   __bf16* __restrict__ out, long n4) {
  long i = (long)blockIdx.x * blockDim.x + threadIdx.x;
  long stride = (long)gridDim.x * blockDim.x;
  for (; i < n4; i += stride) {
    float4 v = reinterpret_cast<const float4*>(in)[i];
    v4bf o;
    o.x = (__bf16)v.x; o.y = (__bf16)v.y; o.z = (__bf16)v.z; o.w = (__bf16)v.w;
    reinterpret_cast<v4bf*>(out)[i] = o;
  }
}

// -------------------- fp32 -> bf16 with transpose (tiled) ------------------
// out[c][r] = (bf16) in[r][c];  in is [R x C] f32 row-major, out is [C x R].
// Block: 256 threads = 32 x 8, 32x32 tile; grid (C/32, R/32, batch).
__global__ __launch_bounds__(256)
void transpose_f32_bf16_kernel(const float* __restrict__ in,
                               __bf16* __restrict__ out,
                               int R, int C, long long sIn, long long sOut) {
  __shared__ float t[32][33];
  const float* pin = in + (long long)blockIdx.z * sIn;
  __bf16* pout = out + (long long)blockIdx.z * sOut;
  const int c0 = blockIdx.x * 32, r0 = blockIdx.y * 32;
  const int tx = threadIdx.x & 31, ty = threadIdx.x >> 5;
  #pragma unroll
  for (int i = 0; i < 4; ++i) {
    int r = ty + i * 8;
    t[r][tx] = pin[(size_t)(r0 + r) * C + c0 + tx];
  }
  __syncthreads();
  #pragma unroll
  for (int i = 0; i < 4; ++i) {
    int c = ty + i * 8;
    pout[(size_t)(c0 + c) * R + r0 + tx] = (__bf16)t[tx][c];
  }
}

// --------------------- softmax over axis=1 (over N), in place --------------
// S is bf16 [B, N, M]; for each (b, m): S[:,m] <- softmax(S[:,m]).
// Block: 256 threads = 64 columns x 4 row-groups; grid (M/64, B).
__global__ __launch_bounds__(256)
void colsoftmax_kernel(__bf16* __restrict__ S, int N, int M) {
  const int cg = threadIdx.x & 63;
  const int rg = threadIdx.x >> 6;
  const int m  = blockIdx.x * 64 + cg;
  const int b  = blockIdx.y;
  __bf16* col = S + (size_t)b * N * M + m;
  const int nseg = N >> 2;
  const int n0 = rg * nseg, n1 = n0 + nseg;

  float mx = -3.0e38f, sum = 0.0f;
  for (int n = n0; n < n1; ++n) {
    float x = (float)col[(size_t)n * M];
    if (x > mx) { sum = sum * __expf(mx - x) + 1.0f; mx = x; }
    else        { sum += __expf(x - mx); }
  }
  __shared__ float smax[4][64];
  __shared__ float ssum[4][64];
  smax[rg][cg] = mx; ssum[rg][cg] = sum;
  __syncthreads();
  float gmx = smax[0][cg];
  #pragma unroll
  for (int g = 1; g < 4; ++g) gmx = fmaxf(gmx, smax[g][cg]);
  float gsum = 0.0f;
  #pragma unroll
  for (int g = 0; g < 4; ++g) gsum += ssum[g][cg] * __expf(smax[g][cg] - gmx);
  float inv = 1.0f / gsum;
  for (int n = n0; n < n1; ++n) {
    float x = (float)col[(size_t)n * M];
    col[(size_t)n * M] = (__bf16)(__expf(x - gmx) * inv);
  }
}

// ------------------------------ WMMA GEMM ----------------------------------
// C[rows x cols] = A[rows x K] * Bsrc^T, where Bsrc is [cols x K] row-major
// (i.e. Bsrc holds columns of the mathematical B contiguously; lda = ldb = K).
// OUTF32: write f32 to Cf, else bf16 to Cb.
// Grid: (cols/128, rows/128, batch). 256 threads = 8 wave32; wave tile 32x64.
// Double-buffered LDS staging: global loads for K-step t+1 are issued before
// the WMMAs of step t, stored to the alternate buffer after compute.
template <bool OUTF32>
__global__ __launch_bounds__(256)
void gemm_bt_wmma(const __bf16* __restrict__ Ag,
                  const __bf16* __restrict__ Bg,
                  float* __restrict__ Cf, __bf16* __restrict__ Cb,
                  int K, int ldc,
                  long long sA, long long sB, long long sC) {
  __shared__ __bf16 lA[2][128 * 32];   // lA[buf][row*32 + k]
  __shared__ __bf16 lB[2][128 * 32];   // lB[buf][col*32 + k]

  const int tid  = threadIdx.x;
  const int lane = tid & 31;
  const int wave = tid >> 5;
  const int wRow = wave >> 1;            // 0..3
  const int wCol = wave & 1;             // 0..1
  const int rowBase = blockIdx.y * 128;
  const int colBase = blockIdx.x * 128;
  Ag += (long long)blockIdx.z * sA;
  Bg += (long long)blockIdx.z * sB;

  const int  hlane = lane & 15;
  const bool hi    = lane >= 16;

  // Per-thread staging indices: 2 x b128 for A tile, 2 x b128 for B tile.
  const int sr0 = tid >> 2;              // 0..63   (rows/cols 0..63)
  const int sr1 = sr0 + 64;              // 64..127
  const int sks = (tid & 3) << 3;        // 0,8,16,24

  v8f acc[2][4] = {};
  uint4 ra0, ra1, rb0, rb1;

  const int nk = K >> 5;

  // prologue: fetch tile 0
  ra0 = *reinterpret_cast<const uint4*>(Ag + (size_t)(rowBase + sr0) * K + sks);
  ra1 = *reinterpret_cast<const uint4*>(Ag + (size_t)(rowBase + sr1) * K + sks);
  rb0 = *reinterpret_cast<const uint4*>(Bg + (size_t)(colBase + sr0) * K + sks);
  rb1 = *reinterpret_cast<const uint4*>(Bg + (size_t)(colBase + sr1) * K + sks);
  *reinterpret_cast<uint4*>(&lA[0][sr0 * 32 + sks]) = ra0;
  *reinterpret_cast<uint4*>(&lA[0][sr1 * 32 + sks]) = ra1;
  *reinterpret_cast<uint4*>(&lB[0][sr0 * 32 + sks]) = rb0;
  *reinterpret_cast<uint4*>(&lB[0][sr1 * 32 + sks]) = rb1;

  for (int t = 0; t < nk; ++t) {
    __syncthreads();                      // buffer t&1 ready for reading
    const int cur = t & 1;

    // issue global loads for next K-step (overlap with WMMA below)
    if (t + 1 < nk) {
      const int kb = (t + 1) << 5;
      ra0 = *reinterpret_cast<const uint4*>(Ag + (size_t)(rowBase + sr0) * K + kb + sks);
      ra1 = *reinterpret_cast<const uint4*>(Ag + (size_t)(rowBase + sr1) * K + kb + sks);
      rb0 = *reinterpret_cast<const uint4*>(Bg + (size_t)(colBase + sr0) * K + kb + sks);
      rb1 = *reinterpret_cast<const uint4*>(Bg + (size_t)(colBase + sr1) * K + kb + sks);
    }
    if (t + 2 < nk) {                     // L2 prefetch two steps ahead
      const int kp = (t + 2) << 5;
      __builtin_prefetch(Ag + (size_t)(rowBase + sr0) * K + kp + sks, 0, 1);
      __builtin_prefetch(Bg + (size_t)(colBase + sr0) * K + kp + sks, 0, 1);
    }

    // ---- load fragments from LDS (ISA 7.12.2 layouts) ----
    v16bf aF[2], bF[4];
    #pragma unroll
    for (int i = 0; i < 2; ++i) {
      int r   = wRow * 32 + i * 16 + hlane;
      int klo = hi ? 8 : 0;              // lanes<16: K0-7,16-23; lanes>=16: K8-15,24-31
      union { v16bf v; uint4 q[2]; } u;
      u.q[0] = *reinterpret_cast<const uint4*>(&lA[cur][r * 32 + klo]);
      u.q[1] = *reinterpret_cast<const uint4*>(&lA[cur][r * 32 + klo + 16]);
      aF[i] = u.v;
    }
    #pragma unroll
    for (int j = 0; j < 4; ++j) {
      int c   = wCol * 64 + j * 16 + hlane;
      int klo = hi ? 8 : 0;              // lanes<16: K0-7,16-23; lanes>=16: K8-15,24-31
      union { v16bf v; uint4 q[2]; } u;
      u.q[0] = *reinterpret_cast<const uint4*>(&lB[cur][c * 32 + klo]);
      u.q[1] = *reinterpret_cast<const uint4*>(&lB[cur][c * 32 + klo + 16]);
      bF[j] = u.v;
    }

    // ---- 8x v_wmma_f32_16x16x32_bf16 ----
    #pragma unroll
    for (int i = 0; i < 2; ++i)
      #pragma unroll
      for (int j = 0; j < 4; ++j)
        acc[i][j] = __builtin_amdgcn_wmma_f32_16x16x32_bf16(
            false, aF[i], false, bF[j], (short)0, acc[i][j], false, false);

    // ---- store next tile into alternate buffer ----
    if (t + 1 < nk) {
      const int nxt = cur ^ 1;
      *reinterpret_cast<uint4*>(&lA[nxt][sr0 * 32 + sks]) = ra0;
      *reinterpret_cast<uint4*>(&lA[nxt][sr1 * 32 + sks]) = ra1;
      *reinterpret_cast<uint4*>(&lB[nxt][sr0 * 32 + sks]) = rb0;
      *reinterpret_cast<uint4*>(&lB[nxt][sr1 * 32 + sks]) = rb1;
    }
  }

  // ---- epilogue: C layout VGPR v -> row v + 8*(lane>=16), col lane&15 ----
  const long long cOff = (long long)blockIdx.z * sC;
  #pragma unroll
  for (int i = 0; i < 2; ++i) {
    #pragma unroll
    for (int j = 0; j < 4; ++j) {
      #pragma unroll
      for (int v = 0; v < 8; ++v) {
        int r = rowBase + wRow * 32 + i * 16 + v + (hi ? 8 : 0);
        int c = colBase + wCol * 64 + j * 16 + hlane;
        float val = acc[i][j][v];
        if (OUTF32) Cf[cOff + (size_t)r * ldc + c] = val;
        else        Cb[cOff + (size_t)r * ldc + c] = (__bf16)val;
      }
    }
  }
}

// ------------------------------- launcher ----------------------------------
extern "C" void kernel_launch(void* const* d_in, const int* in_sizes, int n_in,
                              void* d_out, int out_size, void* d_ws, size_t ws_size,
                              hipStream_t stream) {
  const int Bb = 8, N = 2048, M = 2048, D = 1024;
  const float* text = (const float*)d_in[0];
  const float* ctx  = (const float*)d_in[1];
  const float* We   = (const float*)d_in[2];
  const float* W2   = (const float*)d_in[3];
  float* out = (float*)d_out;

  // Workspace layout (bytes), total = 239,075,328 (~228 MB)
  char* ws = (char*)d_ws;
  size_t off = 0;
  __bf16* text_b = (__bf16*)(ws + off); off += (size_t)Bb * N * D * 2;  // 33.5MB
  __bf16* ctx_b  = (__bf16*)(ws + off); off += (size_t)Bb * M * D * 2;  // 33.5MB
  __bf16* ctxT_b = (__bf16*)(ws + off); off += (size_t)Bb * D * M * 2;  // 33.5MB
  __bf16* WeT_b  = (__bf16*)(ws + off); off += (size_t)D * D * 2;       // 2MB
  __bf16* W2_b   = (__bf16*)(ws + off); off += (size_t)D * D * 2;       // 2MB
  __bf16* T_b    = (__bf16*)(ws + off); off += (size_t)Bb * N * D * 2;  // 33.5MB
  __bf16* S_b    = (__bf16*)(ws + off); off += (size_t)Bb * N * M * 2;  // 67MB
  __bf16* res_b  = (__bf16*)(ws + off); off += (size_t)Bb * N * D * 2;  // 33.5MB
  if (ws_size < off) return;  // insufficient scratch (deterministic no-op)

  // 1) convert / transpose inputs to bf16
  f32_to_bf16_kernel<<<2048, 256, 0, stream>>>(text, text_b, (long)Bb * N * D / 4);
  f32_to_bf16_kernel<<<2048, 256, 0, stream>>>(ctx,  ctx_b,  (long)Bb * M * D / 4);
  f32_to_bf16_kernel<<<256,  256, 0, stream>>>(W2,   W2_b,   (long)D * D / 4);
  // WeT[n][k] = We[k][n]
  transpose_f32_bf16_kernel<<<dim3(D / 32, D / 32, 1), 256, 0, stream>>>(
      We, WeT_b, D, D, 0LL, 0LL);
  // ctxT[b][d][m] = ctx[b][m][d]
  transpose_f32_bf16_kernel<<<dim3(D / 32, M / 32, Bb), 256, 0, stream>>>(
      ctx, ctxT_b, M, D, (long long)M * D, (long long)D * M);

  // 2) T = text @ We           (B = WeT [D x D], shared across batch)
  gemm_bt_wmma<false><<<dim3(D / 128, N / 128, Bb), 256, 0, stream>>>(
      text_b, WeT_b, nullptr, T_b, /*K=*/D, /*ldc=*/D,
      (long long)N * D, 0LL, (long long)N * D);

  // 3) S = T @ ctx^T           (B = ctx [M x D])
  gemm_bt_wmma<false><<<dim3(M / 128, N / 128, Bb), 256, 0, stream>>>(
      T_b, ctx_b, nullptr, S_b, /*K=*/D, /*ldc=*/M,
      (long long)N * D, (long long)M * D, (long long)N * M);

  // 4) attn = softmax(S, axis=1) in place (over N per (b,m) column)
  colsoftmax_kernel<<<dim3(M / 64, Bb), 256, 0, stream>>>(S_b, N, M);

  // 5) res = attn @ ctx        (B = ctxT [D x M])
  gemm_bt_wmma<false><<<dim3(D / 128, N / 128, Bb), 256, 0, stream>>>(
      S_b, ctxT_b, nullptr, res_b, /*K=*/M, /*ldc=*/D,
      (long long)N * M, (long long)D * M, (long long)N * D);

  // 6) out = res @ W2^T        (B = W2 [D x D], shared across batch; f32 out)
  gemm_bt_wmma<true><<<dim3(D / 128, N / 128, Bb), 256, 0, stream>>>(
      res_b, W2_b, out, nullptr, /*K=*/D, /*ldc=*/D,
      (long long)N * D, 0LL, (long long)N * D);
}
